// GNN_CMC_2267742732780
// MI455X (gfx1250) — matrix operand order repaired and checked by reading, not compiled, not run on hardware
//
#include <hip/hip_runtime.h>

typedef __attribute__((ext_vector_type(16))) _Float16 v16h;
typedef __attribute__((ext_vector_type(8)))  float    v8f;

static constexpr int NN  = 25000;   // nodes
static constexpr int NE  = 100000;  // edges
static constexpr int FIN = 32;
static constexpr int EFD = 8;
static constexpr int D   = 64;
static constexpr int GG  = 1000;    // graphs

static constexpr int ETW = 64;      // edges per wave (4 M-tiles -> 4x B reuse)
static constexpr int WPB = 2;       // waves per block

// ---------------------------------------------------------------------------
// zero
__global__ void zero_kernel(float* __restrict__ p, int n) {
    int i = blockIdx.x * blockDim.x + threadIdx.x;
    if (i < n) p[i] = 0.f;
}

// ---------------------------------------------------------------------------
// Generic fused linear: out[m*stride+off+n] = act(bias[n] + extra[m*N+n] + A[m,:]@W[:,n])
// W row-major [K,N].
__global__ void lin_kernel(const float* __restrict__ in, const float* __restrict__ W,
                           const float* __restrict__ bias, const float* __restrict__ extra,
                           float* __restrict__ out, int M, int K, int N,
                           int do_relu, int out_stride, int out_off) {
    int idx = blockIdx.x * blockDim.x + threadIdx.x;
    if (idx >= M * N) return;
    int m = idx / N, n = idx - m * N;
    float acc = bias ? bias[n] : 0.f;
    const float* a = in + (size_t)m * K;
    for (int k = 0; k < K; ++k) acc += a[k] * W[(size_t)k * N + n];
    if (extra) acc += extra[idx];
    if (do_relu) acc = fmaxf(acc, 0.f);
    out[(size_t)m * out_stride + out_off + n] = acc;
}

// ---------------------------------------------------------------------------
// Pack M[(i*64+k), o] = We2[k, i*64+o] into WMMA B-fragment order (fp16).
// Fragment order used here (32x16 16-bit B, contiguous k): lane L holds column n = L&15;
// K_local = ((L>>4)&1)*16 + j  (j = half index 0..15).
// Storage: halves at ((c*4 + nt)*32 + L)*16 + j, c = K-chunk (0..127), nt = col tile.
__global__ void packB_kernel(const float* __restrict__ We2, _Float16* __restrict__ Bf) {
    int idx = blockIdx.x * blockDim.x + threadIdx.x;
    if (idx >= 128 * 4 * 32 * 16) return;
    int j  = idx & 15;
    int L  = (idx >> 4) & 31;
    int nt = (idx >> 9) & 3;
    int c  = idx >> 11;
    int kk = c * 32 + ((L >> 4) & 1) * 16 + j;          // global K index (i*64+k)
    int i  = kk >> 6, k = kk & 63;
    int col = nt * 16 + (L & 15);
    Bf[idx] = (_Float16)We2[(size_t)k * 4096 + i * 64 + col];
}

// Pack Bmat[i,o] = be2[i*64+o] (K=64 over i) into fragment order (2 chunks).
__global__ void packB2_kernel(const float* __restrict__ be2, _Float16* __restrict__ Bf) {
    int idx = blockIdx.x * blockDim.x + threadIdx.x;
    if (idx >= 2 * 4 * 32 * 16) return;
    int j  = idx & 15;
    int L  = (idx >> 4) & 31;
    int nt = (idx >> 9) & 3;
    int c  = idx >> 11;
    int i  = c * 32 + ((L >> 4) & 1) * 16 + j;          // 0..63
    int col = nt * 16 + (L & 15);
    Bf[idx] = (_Float16)be2[i * 64 + col];
}

// ---------------------------------------------------------------------------
// Fused NNConv message kernel. One wave = 64 edges (4 M-tiles x 4 N-tiles).
//   hh = relu(edge_attr @ We1 + be1)          (computed in LDS, fp16)
//   xs = x1[src]                              (gathered to LDS, fp16)
//   msg = (xs outer hh) @ M  +  xs @ Bmat     (WMMA, K=4096 + K=64)
//   agg[dst] += msg                           (f32 global atomics)
// Each B chunk is loaded once per wave and reused by 4 M-tiles -> 4x less
// L2->WGP B traffic than a 16-edge wave.
__global__ __launch_bounds__(WPB * 32) void edge_nnconv_kernel(
    const int* __restrict__ ei, const float* __restrict__ eattr,
    const float* __restrict__ x1, const float* __restrict__ We1,
    const float* __restrict__ be1, const _Float16* __restrict__ Bfrag,
    const _Float16* __restrict__ B2frag, float* __restrict__ agg) {
    __shared__ _Float16 xs[WPB][ETW][D];
    __shared__ _Float16 hh[WPB][ETW][D];
    __shared__ float    ea[WPB][ETW][EFD];
    __shared__ int      srcs[WPB][ETW];
    __shared__ int      dsts[WPB][ETW];

    const int w  = threadIdx.x >> 5;
    const int L  = threadIdx.x & 31;
    const int e0 = (blockIdx.x * WPB + w) * ETW;

    // phase 0: indices + edge attrs -> LDS
    #pragma unroll
    for (int t = 0; t < ETW / 32; ++t) {
        int le = t * 32 + L;
        int e  = e0 + le;
        srcs[w][le] = (e < NE) ? ei[e] : 0;
        dsts[w][le] = (e < NE) ? ei[NE + e] : -1;
    }
    #pragma unroll
    for (int t = 0; t < ETW * EFD / 32; ++t) {
        int flat = t * 32 + L;
        int le = flat >> 3, q = flat & 7;
        int e = e0 + le;
        ea[w][le][q] = (e < NE) ? eattr[(size_t)e * EFD + q] : 0.f;
    }
    __syncthreads();

    // phase 1: edge-hidden MLP + x1 gather -> LDS (fp16)
    for (int t = 0; t < ETW * D / 32; ++t) {     // 128 iterations
        int flat = t * 32 + L;
        int le = flat >> 6, o = flat & 63;
        float acc = be1[o];
        #pragma unroll
        for (int q = 0; q < EFD; ++q) acc += ea[w][le][q] * We1[q * D + o];
        hh[w][le][o] = (_Float16)fmaxf(acc, 0.f);
        int e = e0 + le;
        xs[w][le][o] = (e < NE) ? (_Float16)x1[(size_t)srcs[w][le] * D + o]
                                : (_Float16)0.f;
    }
    __syncthreads();

    // phase 2: WMMA over K = 4096 (128 chunks of 32)
    v8f acc[4][4] = {};                       // [m-tile][n-tile]
    const int mL  = L & 15;
    const int khi = (L >> 4) & 1;

    #pragma unroll 1
    for (int c = 0; c < 128; ++c) {
        const int i    = c >> 1;
        const int koff = (c & 1) * 32;
        const v16h* bp = (const v16h*)(Bfrag + (size_t)c * 4 * 32 * 16);
        v16h b0 = bp[0 * 32 + L];
        v16h b1 = bp[1 * 32 + L];
        v16h b2 = bp[2 * 32 + L];
        v16h b3 = bp[3 * 32 + L];
        #pragma unroll
        for (int mt = 0; mt < 4; ++mt) {
            const int m = mt * 16 + mL;
            const _Float16 xv = xs[w][m][i];
            v16h a;
            #pragma unroll
            for (int j = 0; j < 16; ++j) {
                // A k order used here (contiguous, matches packed B): K = khi*16 + j
                int kj = khi * 16 + j;
                a[j] = xv * hh[w][m][koff + kj];
            }
            acc[mt][0] = __builtin_amdgcn_wmma_f32_16x16x32_f16(false, a, false, b0, (short)0, acc[mt][0], false, false);
            acc[mt][1] = __builtin_amdgcn_wmma_f32_16x16x32_f16(false, a, false, b1, (short)0, acc[mt][1], false, false);
            acc[mt][2] = __builtin_amdgcn_wmma_f32_16x16x32_f16(false, a, false, b2, (short)0, acc[mt][2], false, false);
            acc[mt][3] = __builtin_amdgcn_wmma_f32_16x16x32_f16(false, a, false, b3, (short)0, acc[mt][3], false, false);
        }
    }

    // bias term: xs @ Bmat  (K = 64 over i -> 2 chunks)
    #pragma unroll
    for (int c = 0; c < 2; ++c) {
        const v16h* bp = (const v16h*)(B2frag + (size_t)c * 4 * 32 * 16);
        v16h b0 = bp[0 * 32 + L];
        v16h b1 = bp[1 * 32 + L];
        v16h b2 = bp[2 * 32 + L];
        v16h b3 = bp[3 * 32 + L];
        #pragma unroll
        for (int mt = 0; mt < 4; ++mt) {
            const int m = mt * 16 + mL;
            v16h a;
            #pragma unroll
            for (int j = 0; j < 16; ++j) {
                int kj = khi * 16 + j;
                a[j] = xs[w][m][c * 32 + kj];
            }
            acc[mt][0] = __builtin_amdgcn_wmma_f32_16x16x32_f16(false, a, false, b0, (short)0, acc[mt][0], false, false);
            acc[mt][1] = __builtin_amdgcn_wmma_f32_16x16x32_f16(false, a, false, b1, (short)0, acc[mt][1], false, false);
            acc[mt][2] = __builtin_amdgcn_wmma_f32_16x16x32_f16(false, a, false, b2, (short)0, acc[mt][2], false, false);
            acc[mt][3] = __builtin_amdgcn_wmma_f32_16x16x32_f16(false, a, false, b3, (short)0, acc[mt][3], false, false);
        }
    }

    // phase 3: scatter-add to agg[dst]
    // C/D layout: VGPR r <-> row r (lanes 0-15) / r+8 (lanes 16-31); col = lane&15
    const int col = L & 15;
    #pragma unroll
    for (int mt = 0; mt < 4; ++mt) {
        #pragma unroll
        for (int r = 0; r < 8; ++r) {
            int m2 = mt * 16 + r + khi * 8;
            int e  = e0 + m2;
            if (e < NE) {
                float* drow = agg + (size_t)dsts[w][m2] * D;
                atomicAdd(drow +  0 + col, acc[mt][0][r]);
                atomicAdd(drow + 16 + col, acc[mt][1][r]);
                atomicAdd(drow + 32 + col, acc[mt][2][r]);
                atomicAdd(drow + 48 + col, acc[mt][3][r]);
            }
        }
    }
}

// ---------------------------------------------------------------------------
// GRU single step, torch gate order (r,z,n). Wi/Wh are (3D, D) row-major.
__global__ void gru_kernel(const float* __restrict__ x2, const float* __restrict__ h,
                           const float* __restrict__ Wi, const float* __restrict__ Wh,
                           const float* __restrict__ bi, const float* __restrict__ bh,
                           float* __restrict__ hn) {
    int idx = blockIdx.x * blockDim.x + threadIdx.x;
    if (idx >= NN * D) return;
    int n = idx / D, d = idx - n * D;
    const float* xr = x2 + (size_t)n * D;
    const float* hr = h  + (size_t)n * D;
    float gir = bi[d], giz = bi[D + d], gin = bi[2 * D + d];
    float ghr = bh[d], ghz = bh[D + d], ghn = bh[2 * D + d];
    for (int k = 0; k < D; ++k) {
        float xv = xr[k], hv = hr[k];
        gir += xv * Wi[(size_t)(d)         * D + k];
        giz += xv * Wi[(size_t)(D + d)     * D + k];
        gin += xv * Wi[(size_t)(2 * D + d) * D + k];
        ghr += hv * Wh[(size_t)(d)         * D + k];
        ghz += hv * Wh[(size_t)(D + d)     * D + k];
        ghn += hv * Wh[(size_t)(2 * D + d) * D + k];
    }
    float r  = 1.f / (1.f + __expf(-(gir + ghr)));
    float z  = 1.f / (1.f + __expf(-(giz + ghz)));
    float ng = tanhf(gin + r * ghn);
    hn[idx] = (1.f - z) * ng + z * hr[d];
}

// graph pooling scatter-add
__global__ void pool_kernel(const float* __restrict__ hn, const int* __restrict__ batch,
                            float* __restrict__ pooled) {
    int idx = blockIdx.x * blockDim.x + threadIdx.x;
    if (idx >= NN * D) return;
    int n = idx / D, d = idx - n * D;
    atomicAdd(&pooled[(size_t)batch[n] * D + d], hn[idx]);
}

// ---------------------------------------------------------------------------
extern "C" void kernel_launch(void* const* d_in, const int* in_sizes, int n_in,
                              void* d_out, int out_size, void* d_ws, size_t ws_size,
                              hipStream_t stream) {
    const float* x     = (const float*)d_in[0];
    const int*   ei    = (const int*)  d_in[1];
    const float* eattr = (const float*)d_in[2];
    const int*   batch = (const int*)  d_in[3];
    const float* W0    = (const float*)d_in[4];
    const float* b0    = (const float*)d_in[5];
    const float* We1   = (const float*)d_in[6];
    const float* be1   = (const float*)d_in[7];
    const float* We2   = (const float*)d_in[8];
    const float* be2   = (const float*)d_in[9];
    const float* Wroot = (const float*)d_in[10];
    const float* bconv = (const float*)d_in[11];
    const float* Wi    = (const float*)d_in[12];
    const float* Wh    = (const float*)d_in[13];
    const float* bi    = (const float*)d_in[14];
    const float* bh    = (const float*)d_in[15];
    const float* W11   = (const float*)d_in[16];
    const float* b11   = (const float*)d_in[17];
    const float* W12   = (const float*)d_in[18];
    const float* b12   = (const float*)d_in[19];
    const float* W13   = (const float*)d_in[20];
    const float* b13   = (const float*)d_in[21];
    const float* W21   = (const float*)d_in[22];
    const float* b21   = (const float*)d_in[23];
    const float* W22   = (const float*)d_in[24];
    const float* b22   = (const float*)d_in[25];
    const float* W23   = (const float*)d_in[26];
    const float* b23   = (const float*)d_in[27];
    float* out = (float*)d_out;

    // workspace layout
    float* x1     = (float*)d_ws;
    float* x2     = x1 + (size_t)NN * D;
    float* agg    = x2 + (size_t)NN * D;
    float* hn     = agg + (size_t)NN * D;
    float* pooled = hn + (size_t)NN * D;
    float* t1     = pooled + (size_t)GG * D;
    float* t2     = t1 + (size_t)GG * D;
    float* t3     = t2 + (size_t)GG * D;
    float* t4     = t3 + (size_t)GG * D;
    _Float16* Bfrag  = (_Float16*)(t4 + (size_t)GG * D);
    _Float16* B2frag = Bfrag + (size_t)128 * 4 * 32 * 16;

    const int TPB = 256;
    auto blocks = [](long n, int t) { return (int)((n + t - 1) / t); };

    // zero accumulators (atomics require it every call)
    zero_kernel<<<blocks((long)NN * D, TPB), TPB, 0, stream>>>(agg, NN * D);
    zero_kernel<<<blocks((long)GG * D, TPB), TPB, 0, stream>>>(pooled, GG * D);

    // pre-swizzle B fragments (L2-resident thereafter)
    packB_kernel <<<blocks(128L * 4 * 32 * 16, TPB), TPB, 0, stream>>>(We2, Bfrag);
    packB2_kernel<<<blocks(2L * 4 * 32 * 16, TPB), TPB, 0, stream>>>(be2, B2frag);

    // lin0 + relu
    lin_kernel<<<blocks((long)NN * D, TPB), TPB, 0, stream>>>(
        x, W0, b0, nullptr, x1, NN, FIN, D, 1, D, 0);

    // fused NNConv message + scatter (dominant WMMA kernel)
    const int waves = (NE + ETW - 1) / ETW;            // 1563
    edge_nnconv_kernel<<<(waves + WPB - 1) / WPB, WPB * 32, 0, stream>>>(
        ei, eattr, x1, We1, be1, Bfrag, B2frag, agg);

    // x2 = relu(x1@Wroot + agg + bconv)
    lin_kernel<<<blocks((long)NN * D, TPB), TPB, 0, stream>>>(
        x1, Wroot, bconv, agg, x2, NN, D, D, 1, D, 0);

    // GRU step (h = x1)
    gru_kernel<<<blocks((long)NN * D, TPB), TPB, 0, stream>>>(x2, x1, Wi, Wh, bi, bh, hn);

    // graph pooling
    pool_kernel<<<blocks((long)NN * D, TPB), TPB, 0, stream>>>(hn, batch, pooled);

    // head 1 -> out[:,0]
    lin_kernel<<<blocks((long)GG * D, TPB), TPB, 0, stream>>>(
        pooled, W11, b11, nullptr, t1, GG, D, D, 1, D, 0);
    lin_kernel<<<blocks((long)GG * D, TPB), TPB, 0, stream>>>(
        t1, W12, b12, nullptr, t2, GG, D, D, 1, D, 0);
    lin_kernel<<<blocks(GG, TPB), TPB, 0, stream>>>(
        t2, W13, b13, nullptr, out, GG, D, 1, 0, 2, 0);

    // head 2 -> out[:,1]
    lin_kernel<<<blocks((long)GG * D, TPB), TPB, 0, stream>>>(
        pooled, W21, b21, nullptr, t3, GG, D, D, 1, D, 0);
    lin_kernel<<<blocks((long)GG * D, TPB), TPB, 0, stream>>>(
        t3, W22, b22, nullptr, t4, GG, D, D, 1, D, 0);
    lin_kernel<<<blocks(GG, TPB), TPB, 0, stream>>>(
        t4, W23, b23, nullptr, out, GG, D, 1, 0, 2, 1);
}
